// ISA2D_9740985828042
// MI455X (gfx1250) — matrix operand order repaired and checked by reading, not compiled, not run on hardware
//
#include <hip/hip_runtime.h>

// ---------------------------------------------------------------------------
// MI455X (gfx1250) fused dual SelfAttention2D.
//
// Roofline: ~86 GFLOP of GEMM vs ~0.4 GB of mandatory HBM traffic
// (134MB fp32 in + 134MB fp32 out + 2x 67MB bf16 intermediate) -> HBM-bound
// (~17us at 23.3 TB/s).  Therefore: bf16 WMMA (16x16x32, f32 accum) for all
// GEMMs, full per-row fusion in LDS (input read once, one bf16 intermediate
// between layers), weights pre-packed once into WMMA B-fragment lane order
// (0.65MB bf16 -> L2 resident; 192MB L2 makes per-block rereads free vs HBM).
// ---------------------------------------------------------------------------

typedef __attribute__((ext_vector_type(16))) __bf16 v16bf;
typedef __attribute__((ext_vector_type(8)))  __bf16 v8bf;
typedef __attribute__((ext_vector_type(8)))  float  v8f;
typedef __attribute__((ext_vector_type(4)))  float  v4f;

#define BN_EPS 1e-3f

// ---------------- WMMA helpers (CDNA5 wave32 layouts, CDNA5 ISA) --

__device__ __forceinline__ v8f wmma_bf16(v16bf a, v16bf b, v8f c) {
  // (neg_a, A, neg_b, B, c_mod, C, reuse_a, reuse_b)
  return __builtin_amdgcn_wmma_f32_16x16x32_bf16(false, a, false, b, (short)0, c,
                                                 false, false);
}

// A fragment (16x32 bf16, row-major source [16][ldk] in LDS).
// lanes 0-15: M=lane,   elems 0..15 = K0..15
// lanes16-31: M=lane-16,elems 0..15 = K16..31
__device__ __forceinline__ v16bf afrag(const __bf16* A, int ldk, int lane, int kb) {
  const int m    = lane & 15;
  const int koff = (lane < 16) ? 0 : 16;
  const __bf16* p = A + m * ldk + kb * 32 + koff;
  v8bf lo = *(const v8bf*)p;
  v8bf hi = *(const v8bf*)(p + 8);
  return __builtin_shufflevector(lo, hi, 0, 1, 2, 3, 4, 5, 6, 7,
                                 8, 9, 10, 11, 12, 13, 14, 15);
}

// B fragment from pre-packed weight tile: each lane = 16 contiguous bf16.
__device__ __forceinline__ v16bf bfrag_packed(const __bf16* Wt, int lane) {
  return *(const v16bf*)(Wt + lane * 16);
}

// B fragment computing P^T on the fly from row-major P[16][ldk] in LDS:
// B[k][n] = P[n][k]; lanes 0-15: N=lane, K=kb*32+0..15; lanes 16-31: +16.
__device__ __forceinline__ v16bf bfrag_rowmajorT(const __bf16* P, int ldk, int lane,
                                                 int kb) {
  const int n    = lane & 15;
  const int koff = (lane < 16) ? 0 : 16;
  return *(const v16bf*)(P + n * ldk + kb * 32 + koff);
}

// GEMM: D[16,16-tile nt] = A[16,KB*32] @ Wpacked, KB K-steps.
template <int KB>
__device__ __forceinline__ v8f gemm_packedW(const __bf16* A, int ldk,
                                            const __bf16* W, int nt, int lane) {
  v8f acc = {};
  const __bf16* wt = W + (size_t)nt * KB * 512;
#pragma unroll
  for (int kb = 0; kb < KB; ++kb)
    acc = wmma_bf16(afrag(A, ldk, lane, kb), bfrag_packed(wt + kb * 512, lane), acc);
  return acc;
}

// D (v8f) -> LDS bf16 [16][ldn], +bias +relu (BN scale already folded in W).
// C/D layout: VGPR r, lanes 0-15 -> (M=r, N=lane); lanes 16-31 -> (M=r+8).
__device__ __forceinline__ void store_bias_relu(__bf16* dst, int ldn, int lane,
                                                int nt, v8f d, const float* bias) {
  const int n   = nt * 16 + (lane & 15);
  const int m0  = (lane < 16) ? 0 : 8;
  const float b = bias[n];
#pragma unroll
  for (int r = 0; r < 8; ++r)
    dst[(m0 + r) * ldn + n] = (__bf16)fmaxf(d[r] + b, 0.f);
}

__device__ __forceinline__ void store_plain(__bf16* dst, int ldn, int lane, int nt,
                                            v8f d) {
  const int n  = nt * 16 + (lane & 15);
  const int m0 = (lane < 16) ? 0 : 8;
#pragma unroll
  for (int r = 0; r < 8; ++r)
    dst[(m0 + r) * ldn + n] = (__bf16)d[r];
}

// ---------------- weight pre-pack: fp32 [K][N] -> bf16 WMMA-B tiles ----------
// Tile (nt,kb) holds 512 bf16 in lane order: lane L, elem e ->
//   L<16 : (k = kb*32+e,    n = nt*16+L)
//   L>=16: (k = kb*32+16+e, n = nt*16+L-16)
__global__ void pack_w(const float* __restrict__ W, const float* __restrict__ scale,
                       __bf16* __restrict__ dst, int K, int N) {
  int i = blockIdx.x * blockDim.x + threadIdx.x;
  if (i >= K * N) return;
  int k = i / N, n = i % N;
  float s = scale ? (scale[n] * rsqrtf(1.f + BN_EPS)) : 1.f;
  float v = W[(size_t)k * N + n] * s;
  int kb = k >> 5, kl = k & 31, nt = n >> 4;
  int lv = (kl < 16) ? (n & 15) : ((n & 15) + 16);
  int e  = kl & 15;
  dst[((size_t)(nt * (K >> 5) + kb)) * 512 + lv * 16 + e] = (__bf16)v;
}

// ---------------- fused attention layer ------------------------------------
// LAYER==1: 8192 blocks, one 16-position row (b, h, pw); reads fp32 input,
//           writes bf16 intermediate laid out [bb][hh][ww][ch].
// LAYER==2: 8192 blocks, a PAIR of 8-position rows packed into one M=16 tile
//           (block-diagonal softmax mask); reads bf16 intermediate, writes
//           fp32 output with the final permutation baked into store indices.
template <int LAYER>
__global__ __launch_bounds__(256) void attn_fused(
    const float* __restrict__ xin_f32, const __bf16* __restrict__ xin_bf16,
    const __bf16* __restrict__ w_t1, const __bf16* __restrict__ w_t2,
    const __bf16* __restrict__ w_p1, const __bf16* __restrict__ w_p2,
    const __bf16* __restrict__ w_g, const __bf16* __restrict__ w_r,
    const float* __restrict__ b_t1, const float* __restrict__ b_t2,
    const float* __restrict__ b_p1, const float* __restrict__ b_p2,
    __bf16* __restrict__ y_bf16, float* __restrict__ y_f32) {
  __shared__ __align__(64) __bf16 sX[16][256];
  __shared__ __align__(64) __bf16 sT1[16][128];
  __shared__ __align__(64) __bf16 sP1[16][128];
  __shared__ __align__(64) __bf16 sT2[16][128];
  __shared__ __align__(64) __bf16 sP2[16][128];
  __shared__ __align__(64) __bf16 sGt[128][32];  // G transposed, K-padded to 32
  __shared__ __align__(64) float  sS[16][16];
  __shared__ __align__(64) __bf16 sA[16][32];    // softmax(S), K-padded bf16
  __shared__ __align__(64) __bf16 sO[16][128];

  const int tid  = threadIdx.x;
  const int wave = tid >> 5;
  const int lane = tid & 31;
  const int bx   = blockIdx.x;

  int b = 0, h = 0, pw = 0;             // layer 1 geometry
  int b2 = 0, q2h = 0, q2w = 0, t = 0;  // layer 2 geometry
  if (LAYER == 1) {
    b = bx >> 10; h = (bx >> 3) & 127; pw = bx & 7;
  } else {
    b2 = bx >> 10; q2h = (bx >> 6) & 15; q2w = (bx >> 2) & 15; t = bx & 3;
  }

  // ---- stage 0: gather X row(s) into LDS (bf16) -----------------------------
  {
    const int m  = tid >> 4;
    const int c0 = (tid & 15) << 4;
    if (LAYER == 1) {
      // X[m][ch] = inputs[b, h, m*8+pw, ch]  (attention over w with stride 8)
      const float* src =
          xin_f32 + (((size_t)b * 128 + h) * 128 + (m * 8 + pw)) * 256 + c0;
#pragma unroll
      for (int q = 0; q < 4; ++q) {
        v4f v = *(const v4f*)(src + 4 * q);
#pragma unroll
        for (int i = 0; i < 4; ++i) sX[m][c0 + 4 * q + i] = (__bf16)v[i];
      }
    } else {
      // pair rows p2h=2t+r; position j = attention index; M = r*8+j
      const int r = m >> 3, j = m & 7, p2h = 2 * t + r;
      const __bf16* src =
          xin_bf16 +
          (((((size_t)b2 * 8 + p2h) * 8 + j) * 16 + q2h) * 16 + q2w) * 256 + c0;
      *(v8bf*)&sX[m][c0]     = *(const v8bf*)src;
      *(v8bf*)&sX[m][c0 + 8] = *(const v8bf*)(src + 8);
    }
    if (tid < 128) {  // zero K-padding (k=16..31) of Gt
      v16bf z = {};
      *(v16bf*)&sGt[tid][16] = z;
    }
    if (lane == 0) {  // warm L2->WGP path for this wave's first weight tiles
      __builtin_prefetch(w_t1 + (size_t)wave * 8 * 512, 0, 1);
      __builtin_prefetch(w_p1 + (size_t)wave * 8 * 512, 0, 1);
    }
  }
  __syncthreads();

  // ---- stage 1: T1 = relu(X@tw1'+b), P1 = relu(X@pw1'+b)  (8 N-tiles) ------
  {
    v8f t1 = gemm_packedW<8>(&sX[0][0], 256, w_t1, wave, lane);
    store_bias_relu(&sT1[0][0], 128, lane, wave, t1, b_t1);
    v8f p1 = gemm_packedW<8>(&sX[0][0], 256, w_p1, wave, lane);
    store_bias_relu(&sP1[0][0], 128, lane, wave, p1, b_p1);
  }
  __syncthreads();

  // ---- stage 2: T2, P2, G (G stored transposed for the A@G B-fragment) ----
  {
    v8f t2 = gemm_packedW<4>(&sT1[0][0], 128, w_t2, wave, lane);
    store_bias_relu(&sT2[0][0], 128, lane, wave, t2, b_t2);
    v8f p2 = gemm_packedW<4>(&sP1[0][0], 128, w_p2, wave, lane);
    store_bias_relu(&sP2[0][0], 128, lane, wave, p2, b_p2);
    v8f g = gemm_packedW<8>(&sX[0][0], 256, w_g, wave, lane);
    const int n  = wave * 16 + (lane & 15);
    const int k0 = (lane < 16) ? 0 : 8;
    v8bf pk;
#pragma unroll
    for (int r = 0; r < 8; ++r) pk[r] = (__bf16)g[r];
    *(v8bf*)&sGt[n][k0] = pk;  // Gt[ch][pos]
  }
  __syncthreads();

  // ---- stage 3: scores S = T2 @ P2^T (one 16x16 tile, wave 0) --------------
  if (wave == 0) {
    v8f s = {};
#pragma unroll
    for (int kb = 0; kb < 4; ++kb)
      s = wmma_bf16(afrag(&sT2[0][0], 128, lane, kb),
                    bfrag_rowmajorT(&sP2[0][0], 128, lane, kb), s);
    const int n  = lane & 15;
    const int m0 = (lane < 16) ? 0 : 8;
#pragma unroll
    for (int r = 0; r < 8; ++r) sS[m0 + r][n] = s[r];
  }
  __syncthreads();

  // ---- stage 4: row softmax (regul scale; layer2: block-diagonal mask) -----
  if (tid < 16) {
    const float regul = 0.08838834764831845f;  // sqrt(2/256)
    const int base  = (LAYER == 2) ? (tid & 8) : 0;
    const int width = (LAYER == 2) ? 8 : 16;
    float v[16], mx = -3.0e38f;
    for (int n = 0; n < width; ++n) {
      float x = sS[tid][base + n] * regul;
      v[n] = x;
      mx = fmaxf(mx, x);
    }
    float ssum = 0.f;
    for (int n = 0; n < width; ++n) {
      float e = __expf(v[n] - mx);
      v[n] = e;
      ssum += e;
    }
    const float inv = 1.f / ssum;
    for (int k = 0; k < 32; ++k) sA[tid][k] = (__bf16)0.f;  // mask + K-pad
    for (int n = 0; n < width; ++n) sA[tid][base + n] = (__bf16)(v[n] * inv);
  }
  __syncthreads();

  // ---- stage 5: O = softmax(S) @ G  (K padded to 32, one WMMA / N-tile) ----
  {
    v8f o = {};
    o = wmma_bf16(afrag(&sA[0][0], 32, lane, 0),
                  bfrag_rowmajorT(&sGt[wave * 16][0], 32, lane, 0), o);
    store_plain(&sO[0][0], 128, lane, wave, o);
  }
  __syncthreads();

  // ---- stage 6: Y = O @ rw (16 N-tiles) and scatter -------------------------
#pragma unroll
  for (int half = 0; half < 2; ++half) {
    const int nt = wave + half * 8;
    v8f y = gemm_packedW<4>(&sO[0][0], 128, w_r, nt, lane);
    const int ch = nt * 16 + (lane & 15);
    const int m0 = (lane < 16) ? 0 : 8;
    if (LAYER == 1) {
      const int bb = (b * 8 + pw) * 8 + (h >> 4);
      const int hh = ((h >> 3) & 1) * 8 + (h & 7);
      const size_t base = ((size_t)bb * 16 + hh) * 16 * 256;
#pragma unroll
      for (int r = 0; r < 8; ++r)
        y_bf16[base + (size_t)(m0 + r) * 256 + ch] = (__bf16)y[r];
    } else {
#pragma unroll
      for (int r = 0; r < 8; ++r) {
        const int m = m0 + r;
        const int p2h = 2 * t + (m >> 3), p2w = m & 7;
        const int H = p2h * 16 + q2h, W = q2w * 8 + p2w;
        y_f32[(((size_t)b2 * 128 + H) * 128 + W) * 256 + ch] = y[r];
      }
    }
  }
}

// ---------------------------------------------------------------------------
extern "C" void kernel_launch(void* const* d_in, const int* in_sizes, int n_in,
                              void* d_out, int out_size, void* d_ws, size_t ws_size,
                              hipStream_t stream) {
  (void)in_sizes; (void)n_in; (void)out_size; (void)ws_size;

  const float* inputs = (const float*)d_in[0];
  // param sets: 14 leaves each, dict order:
  // 0 tw1, 1 ts1, 2 tb1, 3 tw2, 4 ts2, 5 tb2,
  // 6 pw1, 7 ps1, 8 pb1, 9 pw2, 10 ps2, 11 pb2, 12 gw, 13 rw
  const float* P1[14];
  const float* P2[14];
  for (int i = 0; i < 14; ++i) P1[i] = (const float*)d_in[1 + i];
  for (int i = 0; i < 14; ++i) P2[i] = (const float*)d_in[15 + i];

  // workspace layout: 12 packed bf16 weights (640KB), then bf16 intermediate
  // [512][16][16][256] (64MB).  Total ~64.7MB.
  char* ws = (char*)d_ws;
  size_t off = 0;
  auto take = [&](size_t elems) {
    __bf16* p = (__bf16*)(ws + off);
    off += elems * sizeof(__bf16);
    return p;
  };
  __bf16* w1[6]; __bf16* w2[6];
  const int WK[6] = {256, 128, 256, 128, 256, 128};   // K of tw1,tw2,pw1,pw2,gw,rw
  const int WN[6] = {128, 128, 128, 128, 128, 256};
  for (int i = 0; i < 6; ++i) w1[i] = take((size_t)WK[i] * WN[i]);
  for (int i = 0; i < 6; ++i) w2[i] = take((size_t)WK[i] * WN[i]);
  off = (off + 255) & ~(size_t)255;
  __bf16* inter = (__bf16*)(ws + off);

  // pack all weights (BN scale folded; gw/rw have no BN)
  const int WSRC[6] = {0, 3, 6, 9, 12, 13};   // weight leaf index in set
  const int SSRC[6] = {1, 4, 7, 10, -1, -1};  // scale leaf index (-1 = none)
  for (int i = 0; i < 6; ++i) {
    int total = WK[i] * WN[i];
    dim3 g((total + 255) / 256);
    pack_w<<<g, 256, 0, stream>>>(P1[WSRC[i]],
                                  SSRC[i] >= 0 ? P1[SSRC[i]] : nullptr, w1[i],
                                  WK[i], WN[i]);
    pack_w<<<g, 256, 0, stream>>>(P2[WSRC[i]],
                                  SSRC[i] >= 0 ? P2[SSRC[i]] : nullptr, w2[i],
                                  WK[i], WN[i]);
  }

  // layer 1: 8192 rows of 16 positions
  attn_fused<1><<<8192, 256, 0, stream>>>(
      inputs, nullptr, w1[0], w1[1], w1[2], w1[3], w1[4], w1[5],
      P1[2], P1[5], P1[8], P1[11], inter, nullptr);

  // layer 2: 8192 blocks, each a pair of 8-position rows
  attn_fused<2><<<8192, 256, 0, stream>>>(
      nullptr, inter, w2[0], w2[1], w2[2], w2[3], w2[4], w2[5],
      P2[2], P2[5], P2[8], P2[11], nullptr, (float*)d_out);
}
